// FastLayer_695784702458
// MI455X (gfx1250) — matrix operand order repaired and checked by reading, not compiled, not run on hardware
//
#include <hip/hip_runtime.h>
#include <hip/hip_bf16.h>
#include <cstdint>

typedef _Float16 h16;
typedef __attribute__((ext_vector_type(16))) _Float16 v16h;
typedef __attribute__((ext_vector_type(8)))  _Float16 v8h;
typedef __attribute__((ext_vector_type(8)))  float    v8f;

#define WMMA_F32_F16(a,b,c) \
  __builtin_amdgcn_wmma_f32_16x16x32_f16(false,(a),false,(b),(short)0,(c),false,false)

constexpr int Bc = 4, Tc = 2048, Dc = 1024, Sc = 64;

// ---- CDNA5 async global->LDS copy (16B per lane, tracked by ASYNCcnt) ----
__device__ inline void async_ld16(uint32_t lds_off, const void* gptr) {
  asm volatile("global_load_async_to_lds_b128 %0, %1, off"
               :: "v"(lds_off), "v"(gptr) : "memory");
}
__device__ inline void async_wait0() {
  asm volatile("s_wait_asynccnt 0" ::: "memory");
}
__device__ inline uint32_t lds_off32(const void* p) {
  return (uint32_t)(uintptr_t)p;   // low 32 bits of generic addr == LDS offset
}

// ---- WMMA operand loaders (CDNA5 16x16x32 f16 layouts, wave32) ----
// A tile: logical 16(M) x 32(K), p -> element [0][0], row stride ld (halves).
// lane: m = lane&15, h = lane>>4; holds K in [8h,8h+8) and [16+8h,16+8h+8).
__device__ inline v16h load_A16x32(const h16* p, int ld, int lane) {
  int m = lane & 15, h = (lane >> 4) & 1;
  const h16* r = p + m * ld + 8 * h;
  v8h lo = *(const v8h*)(r);
  v8h hi = *(const v8h*)(r + 16);
  return __builtin_shufflevector(lo, hi, 0,1,2,3,4,5,6,7,8,9,10,11,12,13,14,15);
}
// B tile: p -> Bt[0][0] where Bt is 16(N) x 32(K) row-major, stride ld.
// lane: n = lane&15, h = lane>>4; holds K in [8h,8h+8) and [16+8h,16+8h+8).
__device__ inline v16h load_B32x16(const h16* p, int ld, int lane) {
  int n = lane & 15, h = (lane >> 4) & 1;
  const h16* r = p + n * ld + 8 * h;
  v8h lo = *(const v8h*)(r);
  v8h hi = *(const v8h*)(r + 16);
  return __builtin_shufflevector(lo, hi, 0,1,2,3,4,5,6,7,8,9,10,11,12,13,14,15);
}

// ---------------- kernel 1: weight conversion ----------------
__global__ __launch_bounds__(256) void prep_weights(
    const float* Wk, const float* Wv, const float* Wq, const float* Wout,
    h16* Wh /*[192][1024]*/, h16* Woh /*[1024][64]*/) {
  int i = blockIdx.x * 256 + threadIdx.x;
  if (i < 192 * 1024) {
    int row = i >> 10, col = i & 1023;
    const float* src = (row < 64) ? Wk : ((row < 128) ? Wv : Wq);
    Wh[i] = (h16)src[(row & 63) * 1024 + col];
  }
  if (i < 1024 * 64) Woh[i] = (h16)Wout[i];
}

// ---------------- kernel 2: projections + RoPE + gates ----------------
__global__ __launch_bounds__(128) void proj_kernel(
    const float* __restrict__ x, const h16* __restrict__ Wh,
    const float* __restrict__ Wg, const float* __restrict__ bg,
    const float* __restrict__ cosT, const float* __restrict__ sinT,
    h16* __restrict__ kr, h16* __restrict__ qr, h16* __restrict__ av,
    float* __restrict__ logg)
{
  __shared__ alignas(16) h16 xs[64][32];
  __shared__ alignas(16) h16 wsh[192][32];
  __shared__ float wgs[32];
  __shared__ float alpha_s[4][16];

  int tid = threadIdx.x, lane = tid & 31, wave = tid >> 5;
  int g0 = blockIdx.x * 64;     // flattened (b*T + t) base
  int t0 = g0 % Tc;             // in-sequence position (for RoPE tables)

  v8f acc[12];
#pragma unroll
  for (int i = 0; i < 12; ++i)
#pragma unroll
    for (int e = 0; e < 8; ++e) acc[i][e] = 0.f;
  float accg = 0.f;

  for (int kk = 0; kk < Dc / 32; ++kk) {
    // async DMA: W tile (f16, no transform needed) -> LDS, 6144 halves
#pragma unroll
    for (int it = 0; it < 6; ++it) {
      int cidx = tid + it * 128;                 // 768 16B chunks
      int row = cidx >> 2, cc = (cidx & 3) * 8;
      async_ld16(lds_off32(&wsh[row][cc]),
                 (const void*)(Wh + (size_t)row * Dc + kk * 32 + cc));
    }
    // manual staging with f32->f16 convert: x tile
    for (int idx = tid; idx < 64 * 32; idx += 128) {
      int r = idx >> 5, c = idx & 31;
      xs[r][c] = (h16)x[(size_t)(g0 + r) * Dc + kk * 32 + c];
    }
    if (tid < 32) wgs[tid] = Wg[kk * 32 + tid];
    async_wait0();
    __syncthreads();

    v16h a = load_A16x32(&xs[16 * wave][0], 32, lane);
#pragma unroll
    for (int ct = 0; ct < 12; ++ct) {
      v16h bb = load_B32x16(&wsh[16 * ct][0], 32, lane);
      acc[ct] = WMMA_F32_F16(a, bb, acc[ct]);
    }
    {   // gate dot-product partial: lane handles row lane&15, half h of K
      int r = lane & 15, h = lane >> 4;
#pragma unroll
      for (int i = 0; i < 16; ++i)
        accg += (float)xs[16 * wave + r][16 * h + i] * wgs[16 * h + i];
    }
    __syncthreads();
  }

  accg += __shfl_xor(accg, 16, 32);
  float alpha = 1.f / (1.f + __expf(-(accg + bg[0])));
  if (lane < 16) alpha_s[wave][lane] = alpha;   // row = lane&15
  __syncthreads();

  int n = lane & 15, h = lane >> 4;
  // RoPE: pair (j, j+32) lives in (tile tl, tile tl+2), same lane/vgpr.
#pragma unroll
  for (int tl = 0; tl < 2; ++tl) {
    int j = tl * 16 + n;        // j < 32
#pragma unroll
    for (int r = 0; r < 8; ++r) {
      int t = t0 + 16 * wave + r + 8 * h;
      float c = cosT[(size_t)t * Sc + j];
      float s = sinT[(size_t)t * Sc + j];
      float x1 = acc[tl][r],     x2 = acc[tl + 2][r];       // k
      acc[tl][r]     = x1 * c - x2 * s;
      acc[tl + 2][r] = x2 * c + x1 * s;
      float y1 = acc[8 + tl][r], y2 = acc[8 + tl + 2][r];   // q
      acc[8 + tl][r]     = y1 * c - y2 * s;
      acc[8 + tl + 2][r] = y2 * c + y1 * s;
    }
  }
  // k-norm, gate, log-decay; scale v by alpha
#pragma unroll
  for (int r = 0; r < 8; ++r) {
    float p = acc[0][r]*acc[0][r] + acc[1][r]*acc[1][r]
            + acc[2][r]*acc[2][r] + acc[3][r]*acc[3][r];
    p += __shfl_xor(p, 1, 32);
    p += __shfl_xor(p, 2, 32);
    p += __shfl_xor(p, 4, 32);
    p += __shfl_xor(p, 8, 32);
    int m = r + 8 * h;
    float a_r = alpha_s[wave][m];
    float z  = a_r * p * 0.1f;
    float gg = 1.f / (1.f + __expf(z));     // exp(-softplus(z)) == sigmoid(-z)
    float lg = __logf(gg + 1e-8f);
    if (n == 0) logg[g0 + 16 * wave + m] = lg;
#pragma unroll
    for (int tl = 0; tl < 4; ++tl) acc[4 + tl][r] *= a_r;   // alpha * v
  }
  // store k_r / alpha_v / q_r as f16 row-major [B*T][64]
#pragma unroll
  for (int tl = 0; tl < 4; ++tl)
#pragma unroll
    for (int r = 0; r < 8; ++r) {
      size_t row = (size_t)(g0 + 16 * wave + r + 8 * h);
      int col = tl * 16 + n;
      kr[row * Sc + col] = (h16)acc[tl][r];
      av[row * Sc + col] = (h16)acc[4 + tl][r];
      qr[row * Sc + col] = (h16)acc[8 + tl][r];
    }
}

// ---------------- kernel 3: per-batch cumsum of log_g ----------------
__global__ __launch_bounds__(256) void scan_kernel(const float* logg, float* cum) {
  __shared__ float part[256];
  __shared__ float excl[256];
  int b = blockIdx.x, tid = threadIdx.x;
  const float* lg = logg + (size_t)b * Tc;
  float* cm = cum + (size_t)b * Tc;
  float v[8], run = 0.f;
#pragma unroll
  for (int i = 0; i < 8; ++i) { run += lg[tid * 8 + i]; v[i] = run; }
  part[tid] = run;
  __syncthreads();
  if (tid == 0) {
    float s = 0.f;
    for (int i = 0; i < 256; ++i) { excl[i] = s; s += part[i]; }
  }
  __syncthreads();
  float off = excl[tid];
#pragma unroll
  for (int i = 0; i < 8; ++i) cm[tid * 8 + i] = off + v[i];
}

// ---------------- kernel 4: decayed attention + past ----------------
__global__ __launch_bounds__(128) void attn_kernel(
    const h16* __restrict__ kr, const h16* __restrict__ qr,
    const h16* __restrict__ av, const float* __restrict__ cum,
    const float* __restrict__ state, float* __restrict__ ret)
{
  __shared__ alignas(16) h16 qs[64][64];
  __shared__ alignas(16) h16 ks[64][64];
  __shared__ alignas(16) h16 avT[64][64];     // [v][s]
  __shared__ alignas(16) h16 wl[64][64];      // decayed attn, row-major [t][s]
  __shared__ alignas(16) h16 ss[64][64];      // state [v][k]
  __shared__ float cshs[64];                  // cum_shift for t-rows
  __shared__ float cums[64];                  // cum for s-rows

  int tid = threadIdx.x, lane = tid & 31, wave = tid >> 5;
  int b = blockIdx.x >> 5;        // 32 t-blocks per batch
  int tb = blockIdx.x & 31;
  int t0 = tb * 64;
  size_t gbase = (size_t)b * Tc;

  // async DMA: q tile (contiguous 8KB) -> LDS
#pragma unroll
  for (int it = 0; it < 4; ++it) {
    int cb = tid + it * 128;                   // 512 16B chunks
    async_ld16(lds_off32(qs) + cb * 16,
               (const void*)((const char*)(qr + (gbase + t0) * Sc) + cb * 16));
  }
  // manual staging with convert: state f32 -> f16
  for (int idx = tid; idx < 64 * 64; idx += 128) {
    int r = idx >> 6, c = idx & 63;
    ss[r][c] = (h16)state[(size_t)b * Sc * Sc + r * Sc + c];
  }
  if (tid < 64) {
    int t = t0 + tid;
    cshs[tid] = (t == 0) ? 0.f : cum[gbase + t - 1];
  }
  async_wait0();
  __syncthreads();

  int n = lane & 15, h = lane >> 4;

  // past[t][v] = exp(cum_shift[t]) * sum_k q_r[t][k] * state[v][k]
  v8f racc[4];
#pragma unroll
  for (int nt = 0; nt < 4; ++nt) {
    v8f p;
#pragma unroll
    for (int e = 0; e < 8; ++e) p[e] = 0.f;
#pragma unroll
    for (int kstep = 0; kstep < 2; ++kstep) {
      v16h a  = load_A16x32(&qs[16 * wave][32 * kstep], 64, lane);
      v16h bb = load_B32x16(&ss[16 * nt][32 * kstep], 64, lane);
      p = WMMA_F32_F16(a, bb, p);
    }
#pragma unroll
    for (int e = 0; e < 8; ++e) {
      int mi = 16 * wave + e + 8 * h;
      p[e] *= __expf(cshs[mi]);
    }
    racc[nt] = p;
  }

  for (int sb = 0; sb <= tb; ++sb) {
    int s0 = sb * 64;
    __syncthreads();
    // async DMA: k tile (contiguous 8KB) -> LDS; overlaps avT transpose below
#pragma unroll
    for (int it = 0; it < 4; ++it) {
      int cb = tid + it * 128;
      async_ld16(lds_off32(ks) + cb * 16,
                 (const void*)((const char*)(kr + (gbase + s0) * Sc) + cb * 16));
    }
    // manual staging with transpose: alpha_v -> avT[v][s]
    for (int idx = tid; idx < 64 * 64; idx += 128) {
      int r = idx >> 6, c = idx & 63;
      avT[c][r] = av[(gbase + s0 + r) * Sc + c];
    }
    if (tid < 64) cums[tid] = cum[gbase + s0 + tid];
    async_wait0();
    __syncthreads();

    // attn = q_r k_r^T, then decay+mask, restage to LDS as A operand
#pragma unroll
    for (int nt = 0; nt < 4; ++nt) {
      v8f w_;
#pragma unroll
      for (int e = 0; e < 8; ++e) w_[e] = 0.f;
#pragma unroll
      for (int kstep = 0; kstep < 2; ++kstep) {
        v16h a  = load_A16x32(&qs[16 * wave][32 * kstep], 64, lane);
        v16h bb = load_B32x16(&ks[16 * nt][32 * kstep], 64, lane);
        w_ = WMMA_F32_F16(a, bb, w_);
      }
#pragma unroll
      for (int e = 0; e < 8; ++e) {
        int mi = 16 * wave + e + 8 * h;
        int si = nt * 16 + n;
        int t = t0 + mi, s = s0 + si;
        float d = (s < t) ? __expf(cshs[mi] - cums[si]) : 0.f; // strict causal
        wl[mi][si] = (h16)(w_[e] * d);
      }
    }
    __syncthreads();
    // intra += decayed_attn @ alpha_v
#pragma unroll
    for (int nt = 0; nt < 4; ++nt)
#pragma unroll
      for (int kstep = 0; kstep < 2; ++kstep) {
        v16h a  = load_A16x32(&wl[16 * wave][32 * kstep], 64, lane);
        v16h bb = load_B32x16(&avT[16 * nt][32 * kstep], 64, lane);
        racc[nt] = WMMA_F32_F16(a, bb, racc[nt]);
      }
  }

  __syncthreads();
#pragma unroll
  for (int nt = 0; nt < 4; ++nt)
#pragma unroll
    for (int e = 0; e < 8; ++e) {
      int mi = 16 * wave + e + 8 * h;
      ret[(gbase + t0 + mi) * Sc + nt * 16 + n] = racc[nt][e];
    }
}

// ---------------- kernel 5: new_state ----------------
__global__ __launch_bounds__(128) void state_kernel(
    const h16* __restrict__ kr, const h16* __restrict__ av,
    const float* __restrict__ cum, const float* __restrict__ state,
    float* __restrict__ ns_out)
{
  __shared__ alignas(16) h16 aT[64][32];   // [v][t] : alpha_v * decay_to_end
  __shared__ alignas(16) h16 kT[64][32];   // [k][t]
  int tid = threadIdx.x, lane = tid & 31, wave = tid >> 5;
  int b = blockIdx.x;
  size_t gbase = (size_t)b * Tc;
  float cl = cum[gbase + Tc - 1];

  v8f acc[4];
#pragma unroll
  for (int nt = 0; nt < 4; ++nt)
#pragma unroll
    for (int e = 0; e < 8; ++e) acc[nt][e] = 0.f;

  for (int kk = 0; kk < Tc / 32; ++kk) {
    __syncthreads();
    for (int idx = tid; idx < 32 * 64; idx += 128) {
      int t = idx >> 6, c = idx & 63;
      size_t row = gbase + kk * 32 + t;
      float d2e = __expf(cl - cum[row]);
      aT[c][t] = (h16)((float)av[row * Sc + c] * d2e);
      kT[c][t] = kr[row * Sc + c];
    }
    __syncthreads();
    v16h a = load_A16x32(&aT[16 * wave][0], 32, lane);
#pragma unroll
    for (int nt = 0; nt < 4; ++nt) {
      v16h bb = load_B32x16(&kT[16 * nt][0], 32, lane);
      acc[nt] = WMMA_F32_F16(a, bb, acc[nt]);
    }
  }
  int n = lane & 15, h = lane >> 4;
  float fd = __expf(cl);
#pragma unroll
  for (int nt = 0; nt < 4; ++nt)
#pragma unroll
    for (int e = 0; e < 8; ++e) {
      int v = 16 * wave + e + 8 * h, k = nt * 16 + n;
      size_t o = (size_t)b * Sc * Sc + (size_t)v * Sc + k;
      ns_out[o] = state[o] * fd + acc[nt][e];
    }
}

// ---------------- kernel 6: output projection ----------------
__global__ __launch_bounds__(128) void out_kernel(
    const float* __restrict__ ret, const h16* __restrict__ Woh,
    const float* __restrict__ bout, float* __restrict__ out)
{
  __shared__ alignas(16) h16 rs[64][64];
  __shared__ alignas(16) h16 wo[128][64];
  int tid = threadIdx.x, lane = tid & 31, wave = tid >> 5;
  int rb = blockIdx.x >> 3, cb = blockIdx.x & 7;
  size_t g0 = (size_t)rb * 64;
  int d0 = cb * 128;

  // async DMA: Wout tile (contiguous 16KB f16) -> LDS
#pragma unroll
  for (int it = 0; it < 8; ++it) {
    int cbk = tid + it * 128;                  // 1024 16B chunks
    async_ld16(lds_off32(wo) + cbk * 16,
               (const void*)((const char*)(Woh + (size_t)d0 * Sc) + cbk * 16));
  }
  // manual staging with convert: retrieved f32 -> f16
  for (int idx = tid; idx < 64 * 64; idx += 128) {
    int r = idx >> 6, c = idx & 63;
    rs[r][c] = (h16)ret[(g0 + r) * Sc + c];
  }
  async_wait0();
  __syncthreads();

  int n = lane & 15, h = lane >> 4;
#pragma unroll
  for (int nt = 0; nt < 8; ++nt) {
    v8f acc;
#pragma unroll
    for (int e = 0; e < 8; ++e) acc[e] = 0.f;
#pragma unroll
    for (int kstep = 0; kstep < 2; ++kstep) {
      v16h a  = load_A16x32(&rs[16 * wave][32 * kstep], 64, lane);
      v16h bb = load_B32x16(&wo[16 * nt][32 * kstep], 64, lane);
      acc = WMMA_F32_F16(a, bb, acc);
    }
    int d = d0 + nt * 16 + n;
    float bo = bout[d];
#pragma unroll
    for (int e = 0; e < 8; ++e) {
      int mi = 16 * wave + e + 8 * h;
      out[(g0 + mi) * Dc + d] = acc[e] + bo;
    }
  }
}

// ---------------- launcher ----------------
extern "C" void kernel_launch(void* const* d_in, const int* in_sizes, int n_in,
                              void* d_out, int out_size, void* d_ws, size_t ws_size,
                              hipStream_t stream) {
  const float* x     = (const float*)d_in[0];
  const float* state = (const float*)d_in[1];
  const float* Wk    = (const float*)d_in[2];
  const float* Wv    = (const float*)d_in[3];
  const float* Wq    = (const float*)d_in[4];
  const float* Wout  = (const float*)d_in[5];
  const float* bout  = (const float*)d_in[6];
  const float* Wg    = (const float*)d_in[7];
  const float* bg    = (const float*)d_in[8];
  const float* cosT  = (const float*)d_in[9];
  const float* sinT  = (const float*)d_in[10];

  float* out = (float*)d_out;                      // (B,T,D)
  float* ns  = out + (size_t)Bc * Tc * Dc;         // (B,S,S)

  char* p = (char*)d_ws;
  h16*   Wh   = (h16*)p;   p += (size_t)192 * 1024 * sizeof(h16);
  h16*   Woh  = (h16*)p;   p += (size_t)1024 * 64 * sizeof(h16);
  h16*   kr   = (h16*)p;   p += (size_t)Bc * Tc * Sc * sizeof(h16);
  h16*   qr   = (h16*)p;   p += (size_t)Bc * Tc * Sc * sizeof(h16);
  h16*   av   = (h16*)p;   p += (size_t)Bc * Tc * Sc * sizeof(h16);
  float* logg = (float*)p; p += (size_t)Bc * Tc * sizeof(float);
  float* cum  = (float*)p; p += (size_t)Bc * Tc * sizeof(float);
  float* ret  = (float*)p; p += (size_t)Bc * Tc * Sc * sizeof(float);

  prep_weights<<<(192 * 1024 + 255) / 256, 256, 0, stream>>>(Wk, Wv, Wq, Wout, Wh, Woh);
  proj_kernel<<<Bc * Tc / 64, 128, 0, stream>>>(x, Wh, Wg, bg, cosT, sinT, kr, qr, av, logg);
  scan_kernel<<<Bc, 256, 0, stream>>>(logg, cum);
  attn_kernel<<<Bc * (Tc / 64), 128, 0, stream>>>(kr, qr, av, cum, state, ret);
  state_kernel<<<Bc, 128, 0, stream>>>(kr, av, cum, state, ns);
  out_kernel<<<(Bc * Tc / 64) * (Dc / 128), 128, 0, stream>>>(ret, Woh, bout, out);
}
